// GraphAttentionLayer_66391604462113
// MI455X (gfx1250) — compile-verified, matrix operand order checked by reading, not run on hardware
//
#include <hip/hip_runtime.h>
#include <hip/hip_bf16.h>

// ---- problem constants (from reference) ----
#define BB   4
#define NN   2048
#define FIN  256
#define HH   4
#define FF   64
#define HF   256   // H*F
#define LOG2E 1.4426950408889634f

typedef __attribute__((ext_vector_type(16))) _Float16 v16h;
typedef __attribute__((ext_vector_type(8)))  _Float16 v8h;
typedef __attribute__((ext_vector_type(8)))  float    v8f;

// -------------------------------------------------------------------------
// Kernel 1: h = x @ W  (per-row), emit:
//   hT[b][head][f][j]  as f16 (j-contiguous -> WMMA B operand loads)
//   esrcT/edstT[b][h][n] as f32, pre-scaled by log2(e)
// -------------------------------------------------------------------------
__global__ __launch_bounds__(256) void gat_gemm_kernel(
    const float* __restrict__ x, const float* __restrict__ W,
    const float* __restrict__ a, _Float16* __restrict__ hT,
    float* __restrict__ esrcT, float* __restrict__ edstT)
{
    __shared__ float xs[FIN];
    __shared__ float hrow[HF];
    __shared__ float hsrow[FF];

    const int blk = blockIdx.x;          // b*N + i
    const int b   = blk >> 11;           // / NN
    const int i   = blk & (NN - 1);
    const int t   = threadIdx.x;

    xs[t] = x[(size_t)blk * FIN + t];
    __syncthreads();

    float acc = 0.f;
#pragma unroll 8
    for (int k = 0; k < FIN; ++k)
        acc = fmaf(xs[k], W[k * HF + t], acc);

    const int head = t >> 6;
    const int f    = t & 63;
    hT[((size_t)((b * HH + head) * FF + f)) * NN + i] = (_Float16)acc;

    hrow[t] = acc;
    __syncthreads();
    if (t < FF)
        hsrow[t] = hrow[t] + hrow[t + 64] + hrow[t + 128] + hrow[t + 192];
    __syncthreads();

    if (t < 2 * HH) {
        const int hh = t >> 1, isdst = t & 1;
        const float* av = a + hh * 2 * FF + isdst * FF;
        float e = 0.f;
        for (int ff = 0; ff < FF; ++ff)
            e = fmaf(hsrow[ff], av[ff], e);
        e *= LOG2E;                       // exp2 domain
        float* dst = isdst ? edstT : esrcT;
        dst[(size_t)(b * HH + hh) * NN + i] = e;
    }
}

// leaky-relu + mask + exp2; masked -> 0 (select after trans op, no exceptions)
__device__ __forceinline__ float pval(int aij, float es, float ed, float mr)
{
    float l = es + ed;
    l = fmaxf(l, 0.2f * l);               // leakyrelu(x) = max(x, 0.2x)
    float p = __builtin_amdgcn_exp2f(l - mr);
    return aij ? p : 0.0f;
}

// -------------------------------------------------------------------------
// Kernel 2: flash-style masked softmax + P@V via v_wmma_f32_16x16x32_f16.
// Block = (b, head, 16-row i-tile); 4 waves.
// Pass 2: each round covers 128 j's; wave w builds the P tile for its own
// 32-wide j slice ONCE (A-fragment layout is f-independent), shares it via
// LDS, then every wave consumes all 4 fragments for its 16-wide f-tile.
// -------------------------------------------------------------------------
__global__ __launch_bounds__(128) void gat_attn_kernel(
    const int* __restrict__ adj, const _Float16* __restrict__ hT,
    const float* __restrict__ esrcT, const float* __restrict__ edstT,
    float* __restrict__ out)
{
    __shared__ float pm[4][16];
    __shared__ float ps[4][16];
    __shared__ float srow[16];
    __shared__ __align__(16) _Float16 pa[4][32][16];   // A fragments, 4 KB

    const int blk   = blockIdx.x;
    const int itile = blk & 127;
    const int head  = (blk >> 7) & 3;
    const int b     = blk >> 9;
    const int i0    = itile * 16;

    const int tid  = threadIdx.x;
    const int wave = tid >> 5;
    const int lane = tid & 31;
    const int r    = lane & 15;
    const int hi   = lane >> 4;           // 0 or 1 (half-wave group)

    const float* ed   = edstT + (size_t)(b * HH + head) * NN;
    const float  es   = esrcT[(size_t)(b * HH + head) * NN + i0 + r];
    const int*   arow = adj + (size_t)(i0 + r) * NN;

    // ---- pass 1a: per-row max over this wave's j chunk ----
    float mloc = -__builtin_inff();
    const int jbeg = wave * (NN / 4);
    for (int jj = jbeg; jj < jbeg + NN / 4; jj += 8) {
        const int j = jj + hi * 4;
        int4   a4 = *(const int4*)(arow + j);
        float4 e4 = *(const float4*)(ed + j);
        float l;
        l = es + e4.x; l = fmaxf(l, 0.2f * l); if (a4.x) mloc = fmaxf(mloc, l);
        l = es + e4.y; l = fmaxf(l, 0.2f * l); if (a4.y) mloc = fmaxf(mloc, l);
        l = es + e4.z; l = fmaxf(l, 0.2f * l); if (a4.z) mloc = fmaxf(mloc, l);
        l = es + e4.w; l = fmaxf(l, 0.2f * l); if (a4.w) mloc = fmaxf(mloc, l);
    }
    mloc = fmaxf(mloc, __shfl_xor(mloc, 16, 32));
    if (lane < 16) pm[wave][r] = mloc;
    __syncthreads();
    const float mr = fmaxf(fmaxf(pm[0][r], pm[1][r]), fmaxf(pm[2][r], pm[3][r]));

    // ---- pass 1b: per-row sum of exp2(l - mr) ----
    float sloc = 0.f;
    for (int jj = jbeg; jj < jbeg + NN / 4; jj += 8) {
        const int j = jj + hi * 4;
        int4   a4 = *(const int4*)(arow + j);
        float4 e4 = *(const float4*)(ed + j);
        sloc += pval(a4.x, es, e4.x, mr);
        sloc += pval(a4.y, es, e4.y, mr);
        sloc += pval(a4.z, es, e4.z, mr);
        sloc += pval(a4.w, es, e4.w, mr);
    }
    sloc += __shfl_xor(sloc, 16, 32);
    if (lane < 16) ps[wave][r] = sloc;
    __syncthreads();
    if (tid < 16) srow[tid] = ps[0][tid] + ps[1][tid] + ps[2][tid] + ps[3][tid];
    __syncthreads();

    // ---- pass 2: P @ V with WMMA ----
    const int f0  = wave * 16;
    const int klo = hi * 8;   // A/B layout: lanes 0-15 K=0..7,16..23; lanes 16-31 K=8..15,24..31
    const _Float16* bptr = hT + ((size_t)((b * HH + head) * FF + f0 + r)) * NN;

    v8f acc0 = {};
    v8f acc1 = {};
    union AU { v16h v; _Float16 e[16]; };
    union BU { v16h v; v8h h[2]; };

    for (int rd = 0; rd < NN / 128; ++rd) {
        // --- produce: this wave's 16x32 P tile at jp ---
        {
            const int jp = rd * 128 + wave * 32;
            const int ja = jp + klo;       // K = klo+0..7
            const int jb = ja + 16;        // K = klo+16..23
            int4 a0 = *(const int4*)(arow + ja);
            int4 a1 = *(const int4*)(arow + ja + 4);
            int4 a2 = *(const int4*)(arow + jb);
            int4 a3 = *(const int4*)(arow + jb + 4);
            float4 e0 = *(const float4*)(ed + ja);
            float4 e1 = *(const float4*)(ed + ja + 4);
            float4 e2 = *(const float4*)(ed + jb);
            float4 e3 = *(const float4*)(ed + jb + 4);

            const int   av[16] = {a0.x, a0.y, a0.z, a0.w, a1.x, a1.y, a1.z, a1.w,
                                  a2.x, a2.y, a2.z, a2.w, a3.x, a3.y, a3.z, a3.w};
            const float ev[16] = {e0.x, e0.y, e0.z, e0.w, e1.x, e1.y, e1.z, e1.w,
                                  e2.x, e2.y, e2.z, e2.w, e3.x, e3.y, e3.z, e3.w};
            AU A;
#pragma unroll
            for (int q = 0; q < 16; ++q)
                A.e[q] = (_Float16)pval(av[q], es, ev[q], mr);
            *(v16h*)(&pa[wave][lane][0]) = A.v;
        }
        __syncthreads();

        // --- consume: all 4 fragments against this wave's f-tile ---
#pragma unroll
        for (int w2 = 0; w2 < 4; ++w2) {
            const int j0 = rd * 128 + w2 * 32;
            v16h Af = *(const v16h*)(&pa[w2][lane][0]);
            BU Bv;
            Bv.h[0] = *(const v8h*)(bptr + j0 + klo);
            Bv.h[1] = *(const v8h*)(bptr + j0 + klo + 16);
            if (w2 & 1)
                acc1 = __builtin_amdgcn_wmma_f32_16x16x32_f16(
                    false, Af, false, Bv.v, (short)0, acc1, false, false);
            else
                acc0 = __builtin_amdgcn_wmma_f32_16x16x32_f16(
                    false, Af, false, Bv.v, (short)0, acc0, false, false);
        }
        __syncthreads();
    }

    // ---- epilogue: divide by row sums; D layout: VGPR q -> M = q + 8*hi ----
#pragma unroll
    for (int q = 0; q < 8; ++q) {
        const int row = q + hi * 8;
        const float s = srow[row];
        const float invs = (s > 0.f) ? __builtin_amdgcn_rcpf(s) : 0.f;
        const float v = (acc0[q] + acc1[q]) * invs;
        out[((size_t)(b * NN + i0 + row)) * HF + head * FF + f0 + r] = v;
    }
}

extern "C" void kernel_launch(void* const* d_in, const int* in_sizes, int n_in,
                              void* d_out, int out_size, void* d_ws, size_t ws_size,
                              hipStream_t stream)
{
    const float* x   = (const float*)d_in[0];
    const int*   adj = (const int*)d_in[1];
    const float* W   = (const float*)d_in[2];
    const float* a   = (const float*)d_in[3];
    float*       out = (float*)d_out;

    char* ws = (char*)d_ws;
    _Float16* hT   = (_Float16*)ws;                                   // B*H*F*N f16 = 4 MB
    float* esrcT   = (float*)(ws + (size_t)BB * HH * FF * NN * 2);    // B*H*N f32
    float* edstT   = esrcT + (size_t)BB * HH * NN;

    gat_gemm_kernel<<<BB * NN, 256, 0, stream>>>(x, W, a, hT, esrcT, edstT);
    gat_attn_kernel<<<BB * HH * (NN / 16), 128, 0, stream>>>(adj, hT, esrcT, edstT, out);
}
